// FlowGuidedDCN_30863634989195
// MI455X (gfx1250) — matrix operand order repaired and checked by reading, not compiled, not run on hardware
//
#include <hip/hip_runtime.h>
#include <hip/hip_bf16.h>

typedef __attribute__((ext_vector_type(16))) _Float16 v16h;
typedef __attribute__((ext_vector_type(8)))  _Float16 v8h;
typedef __attribute__((ext_vector_type(8)))  float    v8f;

namespace {
constexpr int Bn = 8, Cn = 3, Hn = 512, Wn = 512, On = 16, KS = 3, PADn = 1;
constexpr int KK  = KS * KS;                        // 9 taps
constexpr int CK  = Cn * KK;                        // 27 = reduction depth
constexpr int HW  = Hn * Wn;
constexpr int AS  = 40;                             // LDS A-tile row stride (halves), padded from 32
constexpr int WAVES = 8;                            // 256-thread block
constexpr int TILES_PER_WAVE = (Wn / 16) / WAVES;   // 4
}

__global__ __launch_bounds__(256)
void dcn_wmma_kernel(const float* __restrict__ x,
                     const float* __restrict__ offsets,
                     const float* __restrict__ weight,
                     const float* __restrict__ bias,
                     float* __restrict__ out) {
    // Per-wave private A staging tiles: [16 pixels][40 halves] (K padded 32->40 for banking)
    __shared__ __align__(16) _Float16 Alds[WAVES][16 * AS];

    const int bh = blockIdx.x;              // one output row (b, h) per block
    const int b  = bh / Hn;
    const int h  = bh - b * Hn;

    const int tid  = threadIdx.x;
    const int wave = tid >> 5;
    const int lane = tid & 31;
    const int hi   = lane >> 4;             // K-half select in WMMA layouts
    const int nIdx = lane & 15;             // output-channel column (B/C/D layouts)
    const int mrow = lane & 15;             // pixel row (A layout)

    const float* xb   = x       + (size_t)b * Cn * HW;
    const float* offs = offsets + (size_t)b * (2 * KK) * HW;

    _Float16* Aw = &Alds[wave][0];

    // Zero the padded K columns (27..31) once; phase 1 never touches them.
    if (lane < 16) {
        #pragma unroll
        for (int k = CK; k < 32; ++k) Aw[lane * AS + k] = (_Float16)0.f;
    }

    // B fragment: B[k][n] = weight[n][k], weight flat [O=16][C*KK=27], zero-padded to K=32.
    // Layout: lane (n=lane&15, hi): elements 0..7 -> K = 8*hi..8*hi+7, elements 8..15 -> K = 16+8*hi..16+8*hi+7.
    v16h bfrag;
    #pragma unroll
    for (int e = 0; e < 16; ++e) {
        const int k = e + 8 * hi + 8 * (e >> 3);
        bfrag[e] = (k < CK) ? (_Float16)weight[nIdx * CK + k] : (_Float16)0.f;
    }

    // C init: C[m][n] = bias[n]; every accumulator element of a lane has n = nIdx.
    const float bv = bias[nIdx];

    for (int tile = 0; tile < TILES_PER_WAVE; ++tile) {
        const int w0 = (wave * TILES_PER_WAVE + tile) * 16;

        // ---- Phase 1: cooperative bilinear sampling into the LDS A-tile ----
        // 144 tasks = 16 pixels x 9 taps; coordinates computed once per (pixel,tap),
        // reused across the 3 input channels.
        for (int t = lane; t < 16 * KK; t += 32) {
            const int m  = t / KK;
            const int kk = t - m * KK;
            const int w  = w0 + m;
            const int ky = kk / KS;
            const int kx = kk - ky * KS;

            const float offy = offs[((2 * kk    ) * Hn + h) * Wn + w];
            const float offx = offs[((2 * kk + 1) * Hn + h) * Wn + w];
            const float py = offy + (float)(h + ky - PADn);
            const float px = offx + (float)(w + kx - PADn);
            const float y0f = floorf(py);
            const float x0f = floorf(px);
            const float fy = py - y0f;
            const float fx = px - x0f;
            const int y0 = (int)y0f, x0 = (int)x0f;
            const int y1 = y0 + 1,  x1 = x0 + 1;
            // Fold OOB validity into the bilinear weights (branch-free).
            const float vy0 = (y0 >= 0 && y0 < Hn) ? 1.f : 0.f;
            const float vy1 = (y1 >= 0 && y1 < Hn) ? 1.f : 0.f;
            const float vx0 = (x0 >= 0 && x0 < Wn) ? 1.f : 0.f;
            const float vx1 = (x1 >= 0 && x1 < Wn) ? 1.f : 0.f;
            const int yc0 = min(max(y0, 0), Hn - 1);
            const int yc1 = min(max(y1, 0), Hn - 1);
            const int xc0 = min(max(x0, 0), Wn - 1);
            const int xc1 = min(max(x1, 0), Wn - 1);
            const float w00 = (1.f - fy) * (1.f - fx) * vy0 * vx0;
            const float w01 = (1.f - fy) * fx         * vy0 * vx1;
            const float w10 = fy         * (1.f - fx) * vy1 * vx0;
            const float w11 = fy         * fx         * vy1 * vx1;
            const int i00 = yc0 * Wn + xc0;
            const int i01 = yc0 * Wn + xc1;
            const int i10 = yc1 * Wn + xc0;
            const int i11 = yc1 * Wn + xc1;

            _Float16* dst = &Aw[m * AS + kk];
            #pragma unroll
            for (int c = 0; c < Cn; ++c) {
                const float* xp = xb + c * HW;   // x fits in L2; gathers mostly cache hits
                const float s = xp[i00] * w00 + xp[i01] * w01 +
                                xp[i10] * w10 + xp[i11] * w11;
                dst[c * KK] = (_Float16)s;       // A[m][c*9+kk]
            }
        }

        // Wave-private LDS fence (no workgroup barrier needed: LDS ops are
        // in-order per wave; this blocks compiler reordering + waits stores).
        __asm__ volatile("s_wait_dscnt 0" ::: "memory");

        // ---- Phase 2: A fragment from LDS ----
        // Lane (m=lane&15, hi): elements 0..7 -> K = 8*hi..8*hi+7,
        //                       elements 8..15 -> K = 16+8*hi..16+8*hi+7.
        const v8h alo = *reinterpret_cast<const v8h*>(&Aw[mrow * AS + 8 * hi]);
        const v8h ahi = *reinterpret_cast<const v8h*>(&Aw[mrow * AS + 16 + 8 * hi]);
        v16h afrag;
        #pragma unroll
        for (int i = 0; i < 8; ++i) { afrag[i] = alo[i]; afrag[i + 8] = ahi[i]; }

        __asm__ volatile("" ::: "memory");  // keep next tile's DS stores below these loads

        // ---- Phase 3: one WMMA per 16x16 output tile ----
        v8f acc;
        #pragma unroll
        for (int r = 0; r < 8; ++r) acc[r] = bv;
        acc = __builtin_amdgcn_wmma_f32_16x16x32_f16(
                  /*neg_a=*/false, afrag, /*neg_b=*/false, bfrag,
                  /*c_mod=*/(short)0, acc, /*reuse_a=*/false, /*reuse_b=*/false);

        // D element r: out[b][nIdx][h][w0 + r + 8*hi]
        float* op = out + ((size_t)(b * On + nIdx) * Hn + h) * Wn + w0 + 8 * hi;
        #pragma unroll
        for (int r = 0; r < 8; ++r) op[r] = acc[r];
    }
}

extern "C" void kernel_launch(void* const* d_in, const int* in_sizes, int n_in,
                              void* d_out, int out_size, void* d_ws, size_t ws_size,
                              hipStream_t stream) {
    (void)in_sizes; (void)n_in; (void)d_ws; (void)ws_size; (void)out_size;
    const float* x       = (const float*)d_in[0];
    const float* offsets = (const float*)d_in[1];
    const float* weight  = (const float*)d_in[2];
    const float* bias    = (const float*)d_in[3];
    float* out = (float*)d_out;

    dim3 grid(Bn * Hn);   // 4096 blocks: one (b, h) output row each
    dim3 block(WAVES * 32);
    dcn_wmma_kernel<<<grid, block, 0, stream>>>(x, offsets, weight, bias, out);
}
